// GatedGCN_20744692039845
// MI455X (gfx1250) — matrix operand order repaired and checked by reading, not compiled, not run on hardware
//
#include <hip/hip_runtime.h>
#include <hip/hip_bf16.h>

#define NNODES 50000
#define NEDGES 1600000
#define INF_   64
#define H      128
#define L      5
#define G      512
#define C      10
#define H3     384
#define H4     512
#define NTILES (NNODES / 16)   // 3125 exact

typedef __attribute__((ext_vector_type(16))) __bf16 v16bf;
typedef __attribute__((ext_vector_type(8)))  float  v8f;

union AFrag { v16bf v; uint4 u[2]; };

__device__ __forceinline__ __bf16 f2bf(float f) { return (__bf16)f; }

__device__ __forceinline__ v8f wmma_bf16(v16bf a, v16bf b, v8f c) {
    return __builtin_amdgcn_wmma_f32_16x16x32_bf16(false, a, false, b, (short)0, c, false, false);
}

// B fragment loader: packed layout [((kt*NT + nt)*32 + lane)*16 + e], 32B contiguous per lane.
__device__ __forceinline__ v16bf load_bfrag(const __bf16* P, int kt, int nt, int NT, int lane) {
    AFrag f;
    const uint4* p = reinterpret_cast<const uint4*>(P + ((size_t)(((kt * NT + nt) * 32) + lane) << 4));
    f.u[0] = p[0];
    f.u[1] = p[1];
    return f.v;
}

// A fragment from bf16 row-major [.,H]: row = lane&15, K runs {kb..kb+7} and {kb+16..kb+23}, kb=(lane>>4)*8
__device__ __forceinline__ v16bf load_afrag_bf16(const __bf16* src, int row, int kt, int lane) {
    int kb = (lane >> 4) * 8;
    const __bf16* base = src + (size_t)row * H + kt * 32 + kb;
    AFrag f;
    f.u[0] = *reinterpret_cast<const uint4*>(base);
    f.u[1] = *reinterpret_cast<const uint4*>(base + 16);
    return f.v;
}

// A fragment from fp32 row-major [.,H] with on-the-fly bf16 conversion
__device__ __forceinline__ v16bf load_afrag_f32(const float* src, int row, int kt, int lane) {
    int kb = (lane >> 4) * 8;
    const float* base = src + (size_t)row * H + kt * 32 + kb;
    AFrag f;
#pragma unroll
    for (int i = 0; i < 8; ++i) {
        f.v[i]     = f2bf(base[i]);
        f.v[8 + i] = f2bf(base[16 + i]);
    }
    return f.v;
}

__device__ __forceinline__ float sigmoidf_(float x) { return 1.0f / (1.0f + __expf(-x)); }

// ---------------- weight packing (fp32 -> bf16, WMMA B-fragment order) ----------------
__global__ void pack_b_kernel(const float* __restrict__ src, __bf16* __restrict__ dst,
                              int K, int Ncols, int transposed) {
    int idx = blockIdx.x * blockDim.x + threadIdx.x;
    int total = K * Ncols;
    if (idx >= total) return;
    int NT = Ncols >> 4;
    int e = idx & 15;
    int lane = (idx >> 4) & 31;
    int tl = idx >> 9;
    int nt = tl % NT;
    int kt = tl / NT;
    int k = kt * 32 + (lane >> 4) * 8 + (e & 7) + ((e >> 3) << 4);
    int n = nt * 16 + (lane & 15);
    float v = transposed ? src[(size_t)n * K + k] : src[(size_t)k * Ncols + n];
    dst[idx] = f2bf(v);
}

// ---------------- h0 = pad(x) ----------------
__global__ void init_h_kernel(const float* __restrict__ x, float* __restrict__ hF,
                              __bf16* __restrict__ hB) {
    int idx = blockIdx.x * blockDim.x + threadIdx.x;
    if (idx >= NNODES * H) return;
    int n = idx >> 7, f = idx & 127;
    float v = (f < INF_) ? x[n * INF_ + f] : 0.0f;
    hF[idx] = v;
    hB[idx] = f2bf(v);
}

__global__ void zero_f4_kernel(float4* __restrict__ p, int n4) {
    int idx = blockIdx.x * blockDim.x + threadIdx.x;
    if (idx < n4) p[idx] = make_float4(0.f, 0.f, 0.f, 0.f);
}

// ---------------- m = h @ W_l  (bf16 WMMA, bf16 output) ----------------
__global__ void __launch_bounds__(256) msg_gemm_kernel(const __bf16* __restrict__ hB,
                                                       const __bf16* __restrict__ Wp,
                                                       __bf16* __restrict__ mB) {
    int lane = threadIdx.x & 31;
    int tile = blockIdx.x * 8 + (threadIdx.x >> 5);
    if (tile >= NTILES) return;
    int row0 = tile * 16;
    int arow = row0 + (lane & 15);

    v16bf a[4];
#pragma unroll
    for (int kt = 0; kt < 4; ++kt) a[kt] = load_afrag_bf16(hB, arow, kt, lane);

#pragma unroll
    for (int nt = 0; nt < 8; ++nt) {
        v8f acc = {};
#pragma unroll
        for (int kt = 0; kt < 4; ++kt)
            acc = wmma_bf16(a[kt], load_bfrag(Wp, kt, nt, 8, lane), acc);
        int col = nt * 16 + (lane & 15);
        int rb = row0 + (lane >> 4) * 8;
#pragma unroll
        for (int e = 0; e < 8; ++e)
            mB[(size_t)(rb + e) * H + col] = f2bf(acc[e]);
    }
}

// ---------------- gather msg = m[src]*w, scatter-add into agg (L2-resident atomics) ----------------
__global__ void scatter_kernel(const __bf16* __restrict__ mB, const int* __restrict__ srcI,
                               const int* __restrict__ dstI, const float* __restrict__ ew,
                               float* __restrict__ agg) {
    int t = blockIdx.x * blockDim.x + threadIdx.x;
    if (t >= NEDGES * 32) return;
    int e = t >> 5;
    int j = (t & 31) << 2;
    int s = srcI[e];
    int d = dstI[e];
    float w = ew[e];
    const __bf16* mp = mB + (size_t)s * H + j;
    float* ap = agg + (size_t)d * H + j;
    atomicAdd(ap + 0, w * (float)mp[0]);
    atomicAdd(ap + 1, w * (float)mp[1]);
    atomicAdd(ap + 2, w * (float)mp[2]);
    atomicAdd(ap + 3, w * (float)mp[3]);
}

// ---------------- fused GRU cell: gi = agg@Wih^T, gh = h@Whh^T, blend -> h' ----------------
__global__ void __launch_bounds__(256) gru_kernel(float* __restrict__ hF, __bf16* __restrict__ hB,
                                                  const float* __restrict__ agg,
                                                  const __bf16* __restrict__ WihP,
                                                  const __bf16* __restrict__ WhhP,
                                                  const float* __restrict__ b_ih,
                                                  const float* __restrict__ b_hh) {
    int lane = threadIdx.x & 31;
    int tile = blockIdx.x * 8 + (threadIdx.x >> 5);
    if (tile >= NTILES) return;
    int row0 = tile * 16;
    int arow = row0 + (lane & 15);

    v16bf ah[4], am[4];
#pragma unroll
    for (int kt = 0; kt < 4; ++kt) {
        ah[kt] = load_afrag_bf16(hB, arow, kt, lane);
        am[kt] = load_afrag_f32(agg, arow, kt, lane);
    }

    int colL = lane & 15;
    int erow0 = row0 + (lane >> 4) * 8;

#pragma unroll
    for (int ft = 0; ft < 8; ++ft) {
        v8f ir = {}, iz = {}, inn = {}, hr = {}, hz = {}, hn = {};
#pragma unroll
        for (int kt = 0; kt < 4; ++kt) {
            ir  = wmma_bf16(am[kt], load_bfrag(WihP, kt, ft,      24, lane), ir);
            iz  = wmma_bf16(am[kt], load_bfrag(WihP, kt, ft + 8,  24, lane), iz);
            inn = wmma_bf16(am[kt], load_bfrag(WihP, kt, ft + 16, 24, lane), inn);
            hr  = wmma_bf16(ah[kt], load_bfrag(WhhP, kt, ft,      24, lane), hr);
            hz  = wmma_bf16(ah[kt], load_bfrag(WhhP, kt, ft + 8,  24, lane), hz);
            hn  = wmma_bf16(ah[kt], load_bfrag(WhhP, kt, ft + 16, 24, lane), hn);
        }
        int col = ft * 16 + colL;
        float bir = b_ih[col], biz = b_ih[H + col], bin = b_ih[2 * H + col];
        float bhr = b_hh[col], bhz = b_hh[H + col], bhn = b_hh[2 * H + col];
#pragma unroll
        for (int e = 0; e < 8; ++e) {
            size_t o = (size_t)(erow0 + e) * H + col;
            float r = sigmoidf_(ir[e] + bir + hr[e] + bhr);
            float z = sigmoidf_(iz[e] + biz + hz[e] + bhz);
            float n = tanhf(inn[e] + bin + r * (hn[e] + bhn));
            float ho = hF[o];
            float hnew = (1.0f - z) * n + z * ho;
            hF[o] = hnew;
            hB[o] = f2bf(hnew);
        }
    }
}

// ---------------- global max pool over sorted batch ids ----------------
__global__ void pool_init_kernel(float* __restrict__ pooled) {
    int idx = blockIdx.x * blockDim.x + threadIdx.x;
    if (idx < G * H) pooled[idx] = -3.402823466e38f;
}

__global__ void pool_kernel(const float* __restrict__ hF, const int* __restrict__ batch,
                            float* __restrict__ pooled) {
    int idx = blockIdx.x * blockDim.x + threadIdx.x;
    if (idx >= NNODES * H) return;
    int n = idx >> 7, f = idx & 127;
    float v = hF[idx];
    float* addr = pooled + batch[n] * H + f;
    if (v >= 0.0f)
        atomicMax(reinterpret_cast<int*>(addr), __float_as_int(v));
    else
        atomicMin(reinterpret_cast<unsigned int*>(addr), __float_as_uint(v));
}

// ---------------- MLP head (tiny, fp32) ----------------
__global__ void mlp1_kernel(const float* __restrict__ pooled, const float* __restrict__ w,
                            const float* __restrict__ b, float* __restrict__ hid) {
    int idx = blockIdx.x * blockDim.x + threadIdx.x;
    if (idx >= G * H4) return;
    int g = idx >> 9, o = idx & 511;
    const float* pr = pooled + g * H;
    const float* wr = w + (size_t)o * H;
    float s = b[o];
#pragma unroll 4
    for (int k = 0; k < H; ++k) s += pr[k] * wr[k];
    hid[idx] = s > 0.0f ? s : 0.0f;
}

__global__ void mlp2_kernel(const float* __restrict__ hid, const float* __restrict__ w,
                            const float* __restrict__ b, float* __restrict__ out) {
    int idx = blockIdx.x * blockDim.x + threadIdx.x;
    if (idx >= G * C) return;
    int g = idx / C, o = idx % C;
    const float* hr = hid + (size_t)g * H4;
    const float* wr = w + (size_t)o * H4;
    float s = b[o];
#pragma unroll 4
    for (int k = 0; k < H4; ++k) s += hr[k] * wr[k];
    out[idx] = s;
}

extern "C" void kernel_launch(void* const* d_in, const int* in_sizes, int n_in,
                              void* d_out, int out_size, void* d_ws, size_t ws_size,
                              hipStream_t stream) {
    const float* x          = (const float*)d_in[0];
    const int*   edge_index = (const int*)d_in[1];
    const int*   batch      = (const int*)d_in[2];
    const float* edge_w     = (const float*)d_in[3];
    const float* weight     = (const float*)d_in[4];
    const float* w_ih       = (const float*)d_in[5];
    const float* w_hh       = (const float*)d_in[6];
    const float* b_ih       = (const float*)d_in[7];
    const float* b_hh       = (const float*)d_in[8];
    const float* d1w        = (const float*)d_in[9];
    const float* d1b        = (const float*)d_in[10];
    const float* d2w        = (const float*)d_in[11];
    const float* d2b        = (const float*)d_in[12];
    float* out = (float*)d_out;

    char* ws = (char*)d_ws;
    size_t off = 0;
    auto take = [&](size_t bytes) -> void* {
        void* p = ws + off;
        off = (off + bytes + 255) & ~(size_t)255;
        return p;
    };
    float*  hF     = (float*) take((size_t)NNODES * H * 4);
    __bf16* hB     = (__bf16*)take((size_t)NNODES * H * 2);
    __bf16* mB     = (__bf16*)take((size_t)NNODES * H * 2);
    float*  agg    = (float*) take((size_t)NNODES * H * 4);
    __bf16* WlP    = (__bf16*)take((size_t)L * H * H * 2);
    __bf16* WihP   = (__bf16*)take((size_t)H * H3 * 2);
    __bf16* WhhP   = (__bf16*)take((size_t)H * H3 * 2);
    float*  pooled = (float*) take((size_t)G * H * 4);
    float*  hid    = (float*) take((size_t)G * H4 * 4);

    const int* srcI = edge_index;
    const int* dstI = edge_index + NEDGES;

    // Pack weights into WMMA B-fragment order (bf16), once per launch.
    for (int l = 0; l < L; ++l)
        pack_b_kernel<<<(H * H + 255) / 256, 256, 0, stream>>>(weight + (size_t)l * H * H,
                                                               WlP + (size_t)l * H * H, H, H, 0);
    pack_b_kernel<<<(H * H3 + 255) / 256, 256, 0, stream>>>(w_ih, WihP, H, H3, 1);
    pack_b_kernel<<<(H * H3 + 255) / 256, 256, 0, stream>>>(w_hh, WhhP, H, H3, 1);

    init_h_kernel<<<(NNODES * H + 255) / 256, 256, 0, stream>>>(x, hF, hB);

    const int gemm_blocks = (NTILES + 7) / 8;
    for (int l = 0; l < L; ++l) {
        zero_f4_kernel<<<(NNODES * H / 4 + 255) / 256, 256, 0, stream>>>((float4*)agg, NNODES * H / 4);
        msg_gemm_kernel<<<gemm_blocks, 256, 0, stream>>>(hB, WlP + (size_t)l * H * H, mB);
        scatter_kernel<<<(NEDGES * 32) / 256, 256, 0, stream>>>(mB, srcI, dstI, edge_w, agg);
        gru_kernel<<<gemm_blocks, 256, 0, stream>>>(hF, hB, agg, WihP, WhhP, b_ih, b_hh);
    }

    pool_init_kernel<<<(G * H + 255) / 256, 256, 0, stream>>>(pooled);
    pool_kernel<<<(NNODES * H + 255) / 256, 256, 0, stream>>>(hF, batch, pooled);
    mlp1_kernel<<<(G * H4 + 255) / 256, 256, 0, stream>>>(pooled, d1w, d1b, hid);
    mlp2_kernel<<<(G * C + 255) / 256, 256, 0, stream>>>(hid, d2w, d2b, out);
}
